// NEQUIP_2920577761400
// MI455X (gfx1250) — matrix operand order repaired and checked by reading, not compiled, not run on hardware
//
#include <hip/hip_runtime.h>
#include <hip/hip_bf16.h>

typedef __attribute__((ext_vector_type(16))) _Float16 v16h;
typedef __attribute__((ext_vector_type(8)))  _Float16 v8h;
typedef __attribute__((ext_vector_type(8)))  float    v8f;

#define C121  2.1213203435596424f
#define SQRT3 1.7320508075688772f
#define INV_SQRT2 0.7071067811865475f
#define INV_SQRT3 0.5773502691896258f
#define INV_SQRT5 0.4472135954999579f
#define INV_SQRT8 0.35355339059327373f
#define PI_F 3.14159265358979323846f

// swish with hardware v_rcp_f32 (single-op reciprocal) instead of IEEE divide
__device__ __forceinline__ float swishf(float x) {
    return x * __builtin_amdgcn_rcpf(1.0f + __expf(-x));
}

// ---------------------------------------------------------------------------
// Kernel 1: per-node pre-transforms for layer l.
//   s_sc = s @ Wss ; v_sc = einsum(v,Wsv)/sqrt2 ; su = s*Wus ; vu = einsum(v,Wuv)/sqrt2
//   Also zeroes the edge-aggregation buffers for this layer.
// ---------------------------------------------------------------------------
__global__ void node_prep_kernel(int layer, int N,
    const float* __restrict__ nf,
    const float* __restrict__ s_cur_in, const float* __restrict__ v_cur_in,
    const float* __restrict__ Wss, const float* __restrict__ Wsv,
    const float* __restrict__ Wus, const float* __restrict__ Wuv,
    float* __restrict__ su, float* __restrict__ vu,
    float* __restrict__ s_sc, float* __restrict__ v_sc,
    float* __restrict__ agg_s, float* __restrict__ agg_v)
{
    int n = blockIdx.x * blockDim.x + threadIdx.x;
    if (n >= N) return;

    float s;
    float v[2][3];
    if (layer == 0) {
        s = nf[n * 7];
        #pragma unroll
        for (int m = 0; m < 2; ++m)
            #pragma unroll
            for (int c = 0; c < 3; ++c)
                v[m][c] = nf[n * 7 + 1 + m * 3 + c];
    } else {
        s = s_cur_in[n];
        #pragma unroll
        for (int m = 0; m < 2; ++m)
            #pragma unroll
            for (int c = 0; c < 3; ++c)
                v[m][c] = v_cur_in[n * 6 + m * 3 + c];
    }

    su[n] = s * Wus[layer];
    #pragma unroll
    for (int k = 0; k < 2; ++k) {
        float w0 = Wuv[layer * 4 + 0 * 2 + k];
        float w1 = Wuv[layer * 4 + 1 * 2 + k];
        #pragma unroll
        for (int c = 0; c < 3; ++c)
            vu[n * 6 + k * 3 + c] = (v[0][c] * w0 + v[1][c] * w1) * INV_SQRT2;
    }
    #pragma unroll
    for (int k = 0; k < 3; ++k)
        s_sc[n * 3 + k] = s * Wss[layer * 3 + k];
    #pragma unroll
    for (int k = 0; k < 2; ++k) {
        float w0 = Wsv[layer * 4 + 0 * 2 + k];
        float w1 = Wsv[layer * 4 + 1 * 2 + k];
        #pragma unroll
        for (int c = 0; c < 3; ++c)
            v_sc[n * 6 + k * 3 + c] = (v[0][c] * w0 + v[1][c] * w1) * INV_SQRT2;
    }
    #pragma unroll
    for (int j = 0; j < 3; ++j)  agg_s[n * 3 + j] = 0.0f;
    #pragma unroll
    for (int j = 0; j < 15; ++j) agg_v[n * 15 + j] = 0.0f;
}

// ---------------------------------------------------------------------------
// Kernel 2: edge kernel. Each wave handles 32 edges:
//   - recompute radial(8) per edge (bessel * envelope), stage to LDS as f16
//   - radial -> 64 -> 64 -> 8 MLP via v_wmma_f32_16x16x32_f16 (2 tiles of 16 edges)
//   - per-edge tensor-product messages, scaled by mix, atomically scattered.
// Weight B-tiles (f16, ISA B-layout: lane = N column, packed K halves) live in
// registers; inter-GEMM D->A transposes go through per-wave LDS buffers.
// ---------------------------------------------------------------------------
__global__ void __launch_bounds__(256) edge_kernel(int layer, int E,
    const float* __restrict__ positions,
    const int*   __restrict__ senders,
    const int*   __restrict__ receivers,
    const float* __restrict__ W1, const float* __restrict__ W2, const float* __restrict__ W3,
    const float* __restrict__ su, const float* __restrict__ vu,
    float* __restrict__ agg_s, float* __restrict__ agg_v)
{
    // weights transposed to f16: Wt[n][k]
    __shared__ alignas(64) _Float16 sW1t[64][32];   // K=8 valid, zero-padded to 32
    __shared__ alignas(64) _Float16 sW2t[64][64];
    __shared__ alignas(64) _Float16 sW3t[16][64];   // N=8 valid, zero-padded to 16
    // per-wave scratch; sRAD rows 32..47 are a permanent zero pad so that
    // lanes 16..31 can build their (all-zero) A-halves with a uniform load.
    __shared__ alignas(64) _Float16 sRAD[8][48][8];
    __shared__ alignas(64) _Float16 sH1[8][16][64];
    __shared__ alignas(64) _Float16 sH2[8][16][64];
    __shared__ alignas(64) float    sMIX[8][32][8];

    const int tid  = threadIdx.x;
    const int lane = tid & 31;
    const int wid  = tid >> 5;

    const float* W1l = W1 + layer * 8 * 64;
    const float* W2l = W2 + layer * 64 * 64;
    const float* W3l = W3 + layer * 64 * 8;

    for (int i = tid; i < 64 * 32; i += blockDim.x) {
        int n = i >> 5, k = i & 31;
        sW1t[n][k] = (k < 8) ? (_Float16)W1l[k * 64 + n] : (_Float16)0.0f;
    }
    for (int i = tid; i < 64 * 64; i += blockDim.x) {
        int n = i >> 6, k = i & 63;
        sW2t[n][k] = (_Float16)W2l[k * 64 + n];
    }
    for (int i = tid; i < 16 * 64; i += blockDim.x) {
        int n = i >> 6, k = i & 63;
        sW3t[n][k] = (n < 8) ? (_Float16)W3l[k * 8 + n] : (_Float16)0.0f;
    }
    // zero pad rows of this wave's RAD buffer (wave-local, no barrier needed)
    {
        _Float16* pad = &sRAD[wid][32][0];
        for (int i = lane; i < 16 * 8; i += 32) pad[i] = (_Float16)0.0f;
    }
    __syncthreads();

    // ---- hoist B tiles into registers (B layout: lane holds column n = lane&15,
    //      16 consecutive K halves at k0 = (lane<16 ? 0 : 16) + 32*kt) ----
    const int bn = lane & 15;
    const int bk = (lane < 16) ? 0 : 16;
    v16h B1[4], B2[2][4], B3[2];
    #pragma unroll
    for (int nt = 0; nt < 4; ++nt)
        B1[nt] = *(const v16h*)&sW1t[nt * 16 + bn][bk];
    #pragma unroll
    for (int kt = 0; kt < 2; ++kt)
        #pragma unroll
        for (int nt = 0; nt < 4; ++nt)
            B2[kt][nt] = *(const v16h*)&sW2t[nt * 16 + bn][kt * 32 + bk];
    #pragma unroll
    for (int kt = 0; kt < 2; ++kt)
        B3[kt] = *(const v16h*)&sW3t[bn][kt * 32 + bk];

    const int cn  = lane & 15;             // D-layout column
    const int mhi = (lane >= 16) ? 8 : 0;  // D-layout row offset

    const int ntiles = (E + 31) >> 5;
    const int gw = blockIdx.x * (blockDim.x >> 5) + wid;
    const int nw = gridDim.x * (blockDim.x >> 5);

    for (int tile = gw; tile < ntiles; tile += nw) {
        const int  e     = tile * 32 + lane;
        const bool valid = (e < E);
        const int  ec    = valid ? e : (E - 1);
        const int  si    = senders[ec];
        const int  ri    = receivers[ec];

        // ---- geometry + radial basis (one edge per lane) ----
        const float vx = positions[3 * ri + 0] - positions[3 * si + 0];
        const float vy = positions[3 * ri + 1] - positions[3 * si + 1];
        const float vz = positions[3 * ri + 2] - positions[3 * si + 2];
        const float len  = sqrtf(vx * vx + vy * vy + vz * vz);
        const float safe = (len == 0.0f) ? 1.0f : len;
        const float rinv = __builtin_amdgcn_rcpf(safe);
        const float nx = vx * rinv, ny = vy * rinv, nz = vz * rinv;
        const float x  = len;
        const float x2 = x * x, x3 = x2 * x, x6 = x3 * x3;
        const float poly = 1.0f - 28.0f * x6 + 48.0f * x6 * x - 21.0f * x6 * x2;
        const float env  = (x < 1.0f) ? poly : 0.0f;
        const float coef = (len == 0.0f || !valid) ? 0.0f
                                                   : 1.4142135623730951f * env * rinv;
        #pragma unroll
        for (int k = 0; k < 8; ++k)
            sRAD[wid][lane][k] = (_Float16)(coef * __sinf(PI_F * (float)(k + 1) * x));

        // ---- MLP for the two 16-edge tiles ----
        #pragma unroll
        for (int t = 0; t < 2; ++t) {
            // A layout (16x32 f16): lane holds row m=lane&15; lanes<16 carry
            // K=0..15 (8..15 zero), lanes>=16 carry K=16..31 (zero here) -> read zero-pad row.
            const int arow = (lane < 16) ? (t * 16 + cn) : (32 + cn);
            const v8h ra = *(const v8h*)&sRAD[wid][arow][0];
            v16h A;
            #pragma unroll
            for (int j = 0; j < 8; ++j) { A[j] = ra[j]; A[8 + j] = (_Float16)0.0f; }

            // GEMM1: radial @ W1 / sqrt8 -> swish -> H1 (16x64 f16)
            #pragma unroll
            for (int nt = 0; nt < 4; ++nt) {
                v8f c = {};
                c = __builtin_amdgcn_wmma_f32_16x16x32_f16(
                        false, A, false, B1[nt], (short)0, c, false, false);
                #pragma unroll
                for (int r = 0; r < 8; ++r) {
                    float pre = c[r] * INV_SQRT8;
                    sH1[wid][mhi + r][nt * 16 + cn] = (_Float16)swishf(pre);
                }
            }

            // A2 tiles from H1 (A layout: lane = row m, 8+8 consecutive K halves)
            v16h A2[2];
            #pragma unroll
            for (int kt = 0; kt < 2; ++kt) {
                const int k0 = kt * 32 + ((lane < 16) ? 0 : 16);
                const v8h lo = *(const v8h*)&sH1[wid][cn][k0];
                const v8h hi = *(const v8h*)&sH1[wid][cn][k0 + 8];
                #pragma unroll
                for (int j = 0; j < 8; ++j) { A2[kt][j] = lo[j]; A2[kt][8 + j] = hi[j]; }
            }

            // GEMM2: H1 @ W2 / 8 -> swish -> H2 (16x64 f16)
            #pragma unroll
            for (int nt = 0; nt < 4; ++nt) {
                v8f c = {};
                c = __builtin_amdgcn_wmma_f32_16x16x32_f16(
                        false, A2[0], false, B2[0][nt], (short)0, c, false, false);
                c = __builtin_amdgcn_wmma_f32_16x16x32_f16(
                        false, A2[1], false, B2[1][nt], (short)0, c, false, false);
                #pragma unroll
                for (int r = 0; r < 8; ++r) {
                    float pre = c[r] * 0.125f;
                    sH2[wid][mhi + r][nt * 16 + cn] = (_Float16)swishf(pre);
                }
            }

            // A3 tiles from H2
            v16h A3[2];
            #pragma unroll
            for (int kt = 0; kt < 2; ++kt) {
                const int k0 = kt * 32 + ((lane < 16) ? 0 : 16);
                const v8h lo = *(const v8h*)&sH2[wid][cn][k0];
                const v8h hi = *(const v8h*)&sH2[wid][cn][k0 + 8];
                #pragma unroll
                for (int j = 0; j < 8; ++j) { A3[kt][j] = lo[j]; A3[kt][8 + j] = hi[j]; }
            }

            // GEMM3: H2 @ W3 / 8 -> mix (16x8 f32)
            {
                v8f c = {};
                c = __builtin_amdgcn_wmma_f32_16x16x32_f16(
                        false, A3[0], false, B3[0], (short)0, c, false, false);
                c = __builtin_amdgcn_wmma_f32_16x16x32_f16(
                        false, A3[1], false, B3[1], (short)0, c, false, false);
                if (cn < 8) {
                    #pragma unroll
                    for (int r = 0; r < 8; ++r)
                        sMIX[wid][t * 16 + mhi + r][cn] = c[r] * 0.125f;
                }
            }
        }

        // ---- per-edge tensor-product messages + scatter (one edge per lane) ----
        if (valid) {
            float mx[8];
            #pragma unroll
            for (int j = 0; j < 8; ++j) mx[j] = sMIX[wid][lane][j];

            const float se  = su[si];
            const float v0x = vu[si * 6 + 0], v0y = vu[si * 6 + 1], v0z = vu[si * 6 + 2];
            const float v1x = vu[si * 6 + 3], v1y = vu[si * 6 + 4], v1z = vu[si * 6 + 5];
            const float d0 = v0x * nx + v0y * ny + v0z * nz;
            const float d1 = v1x * nx + v1y * ny + v1z * nz;

            atomicAdd(&agg_s[ri * 3 + 0], se * mx[0]);
            atomicAdd(&agg_s[ri * 3 + 1], d0 * mx[1]);
            atomicAdd(&agg_s[ri * 3 + 2], d1 * mx[2]);

            float* av = &agg_v[ri * 15];
            // row0: ve0 * mix3
            atomicAdd(&av[0], v0x * mx[3]);
            atomicAdd(&av[1], v0y * mx[3]);
            atomicAdd(&av[2], v0z * mx[3]);
            // row1: ve1 * mix4
            atomicAdd(&av[3], v1x * mx[4]);
            atomicAdd(&av[4], v1y * mx[4]);
            atomicAdd(&av[5], v1z * mx[4]);
            // row2: pc0 = C121*(d0*n - ve0/3), * mix5
            atomicAdd(&av[6], C121 * (d0 * nx - v0x * (1.0f / 3.0f)) * mx[5]);
            atomicAdd(&av[7], C121 * (d0 * ny - v0y * (1.0f / 3.0f)) * mx[5]);
            atomicAdd(&av[8], C121 * (d0 * nz - v0z * (1.0f / 3.0f)) * mx[5]);
            // row3: pc1, * mix6
            atomicAdd(&av[9],  C121 * (d1 * nx - v1x * (1.0f / 3.0f)) * mx[6]);
            atomicAdd(&av[10], C121 * (d1 * ny - v1y * (1.0f / 3.0f)) * mx[6]);
            atomicAdd(&av[11], C121 * (d1 * nz - v1z * (1.0f / 3.0f)) * mx[6]);
            // row4: pb = sqrt3*se*n, * mix7
            atomicAdd(&av[12], SQRT3 * se * nx * mx[7]);
            atomicAdd(&av[13], SQRT3 * se * ny * mx[7]);
            atomicAdd(&av[14], SQRT3 * se * nz * mx[7]);
        }
    }
}

// ---------------------------------------------------------------------------
// Kernel 3: per-node post-transform + gated update for layer l.
// ---------------------------------------------------------------------------
__global__ void node_update_kernel(int layer, int N,
    const float* __restrict__ Wds, const float* __restrict__ Wdv,
    const float* __restrict__ agg_s, const float* __restrict__ agg_v,
    const float* __restrict__ s_sc, const float* __restrict__ v_sc,
    float* __restrict__ s_cur, float* __restrict__ v_cur,
    float* __restrict__ out, int write_out)
{
    int n = blockIdx.x * blockDim.x + threadIdx.x;
    if (n >= N) return;

    float st[3];
    #pragma unroll
    for (int k = 0; k < 3; ++k) {
        float sd = 0.0f;
        #pragma unroll
        for (int j = 0; j < 3; ++j)
            sd += agg_s[n * 3 + j] * Wds[layer * 9 + j * 3 + k];
        st[k] = sd * INV_SQRT3 + s_sc[n * 3 + k];
    }
    float g0 = swishf(st[0]);
    float g1 = swishf(st[1]);
    float g2 = swishf(st[2]);
    s_cur[n] = g0;

    #pragma unroll
    for (int k = 0; k < 2; ++k) {
        const float gate = (k == 0) ? g1 : g2;
        #pragma unroll
        for (int c = 0; c < 3; ++c) {
            float vd = 0.0f;
            #pragma unroll
            for (int m = 0; m < 5; ++m)
                vd += agg_v[n * 15 + m * 3 + c] * Wdv[layer * 10 + m * 2 + k];
            float vt = vd * INV_SQRT5 + v_sc[n * 6 + k * 3 + c];
            float vn = vt * gate;
            v_cur[n * 6 + k * 3 + c] = vn;
            if (write_out && k == 0)
                out[n * 3 + c] = vn;
        }
    }
}

// ---------------------------------------------------------------------------
extern "C" void kernel_launch(void* const* d_in, const int* in_sizes, int n_in,
                              void* d_out, int out_size, void* d_ws, size_t ws_size,
                              hipStream_t stream)
{
    const float* positions = (const float*)d_in[0];
    const float* node_feat = (const float*)d_in[1];
    const int*   senders   = (const int*)d_in[2];
    const int*   receivers = (const int*)d_in[3];
    const float* Wss = (const float*)d_in[4];
    const float* Wsv = (const float*)d_in[5];
    const float* Wus = (const float*)d_in[6];
    const float* Wuv = (const float*)d_in[7];
    const float* W1  = (const float*)d_in[8];
    const float* W2  = (const float*)d_in[9];
    const float* W3  = (const float*)d_in[10];
    const float* Wds = (const float*)d_in[11];
    const float* Wdv = (const float*)d_in[12];

    const int N = in_sizes[0] / 3;   // positions: (N,3)
    const int E = in_sizes[2];       // senders:   (E,)

    // workspace layout (floats): 41*N total (~8.2 MB for N=50k)
    float* ws    = (float*)d_ws;
    float* su    = ws;               // N
    float* vu    = su    + (size_t)N;        // 6N
    float* s_sc  = vu    + (size_t)6 * N;    // 3N
    float* v_sc  = s_sc  + (size_t)3 * N;    // 6N
    float* agg_s = v_sc  + (size_t)6 * N;    // 3N
    float* agg_v = agg_s + (size_t)3 * N;    // 15N
    float* s_cur = agg_v + (size_t)15 * N;   // N
    float* v_cur = s_cur + (size_t)N;        // 6N
    float* out   = (float*)d_out;

    const int nodeBlocks = (N + 255) / 256;
    const int edgeBlocks = 1024;

    for (int l = 0; l < 3; ++l) {
        node_prep_kernel<<<nodeBlocks, 256, 0, stream>>>(
            l, N, node_feat, s_cur, v_cur, Wss, Wsv, Wus, Wuv,
            su, vu, s_sc, v_sc, agg_s, agg_v);

        edge_kernel<<<edgeBlocks, 256, 0, stream>>>(
            l, E, positions, senders, receivers, W1, W2, W3,
            su, vu, agg_s, agg_v);

        node_update_kernel<<<nodeBlocks, 256, 0, stream>>>(
            l, N, Wds, Wdv, agg_s, agg_v, s_sc, v_sc,
            s_cur, v_cur, out, (l == 2) ? 1 : 0);
    }
}
